// MyLoss_max_62929860821446
// MI455X (gfx1250) — compile-verified, matrix operand order checked by reading, not run on hardware
//
#include <hip/hip_runtime.h>

// ---------------------------------------------------------------------------
// MyLoss_max on MI455X (gfx1250, wave32).
//
// Block = (b, oy, half-row): stages 128ch x 5rows x 33cols of feature into LDS
// once (f16, channel-contiguous, coalesced f32 loads along x, packed-pair
// ds_store_b32). Each wave computes up to 4 output positions: Gram = P*P^T
// (25x128 padded to 32x128) as 2x2 tiles of v_wmma_f32_16x16x32_f16 over 4
// K-chunks (f32 accumulate). Cosine/mean/max reductions stay in LDS per wave;
// per-batch max via atomicMax on the uint bit pattern (values >= 0).
// ---------------------------------------------------------------------------

typedef __attribute__((ext_vector_type(16))) _Float16 v16h;
typedef __attribute__((ext_vector_type(2)))  _Float16 v2h;
typedef __attribute__((ext_vector_type(8)))  float    v8f;

#define KK    5
#define SS    2
#define CCH   128
#define HH    63
#define WW    63
#define OH    30
#define OW    30
#define BB    32
#define XHALF 15            // output positions per block (half row)
#define NX    33            // input columns needed: 2*14+5 = 33
#define XPAD  34
#define CPAD  132           // f16 channel pad: 264B row stride -> bank-spread
#define WAVES 4             // 128 threads
#define NBLK  (BB * OH * 2) // 1920

__global__ __launch_bounds__(WAVES * 32)
void gram_cosine_kernel(const float* __restrict__ feat,
                        unsigned int* __restrict__ bmax)
{
    __shared__ _Float16 stage[KK][XPAD][CPAD];  // 44,880 B
    __shared__ float    gram [WAVES][32][32];   // 16,384 B
    __shared__ float    nrm  [WAVES][32];
    __shared__ float    srow [WAVES][32];

    const int tid     = threadIdx.x;
    const int lane    = tid & 31;
    const int wv      = tid >> 5;
    const int b       = blockIdx.x / (OH * 2);
    const int rem     = blockIdx.x % (OH * 2);
    const int oy      = rem >> 1;
    const int half_id = rem & 1;
    const int y0      = oy * SS;
    const int x0g     = half_id * (XHALF * SS);   // 0 or 30; x0g+32 <= 62

    // ---- stage: coalesced f32 loads along x, packed f16 pair stores --------
    // Elements per channel: 5 rows x 33 cols = 165; 128 threads sweep it.
    const float* fb = feat + (size_t)b * CCH * HH * WW;
    for (int c = 0; c < CCH; c += 2) {
        const float* f0 = fb + (size_t)(c    ) * HH * WW;
        const float* f1 = fb + (size_t)(c + 1) * HH * WW;
        #pragma unroll
        for (int t = 0; t < 2; ++t) {
            const int idx = t * 128 + tid;
            if (idx < KK * NX) {
                const int iy = idx / NX;
                const int ix = idx % NX;
                const size_t g = (size_t)(y0 + iy) * WW + (x0g + ix);
                v2h pk;
                pk[0] = (_Float16)f0[g];
                pk[1] = (_Float16)f1[g];
                *(v2h*)&stage[iy][ix][c] = pk;   // 4B-aligned (c even, CPAD even)
            }
        }
    }
    __syncthreads();

    const int lhalf = lane >> 4;    // 0: lanes 0-15, 1: lanes 16-31
    const int mrow  = lane & 15;

    // ---- each wave: up to 4 positions (pslot = wv + 4*s) -------------------
    for (int s = 0; s < 4; ++s) {
        const int  pslot  = wv + 4 * s;          // 0..15
        const bool active = (pslot < XHALF);
        const int  xloc   = 2 * pslot;           // local x of patch origin

        if (active) {
            // 2x2 tiles, 4 K-chunks of 32 => 16 v_wmma per position
            #pragma unroll
            for (int mt = 0; mt < 2; ++mt) {
                // A fragments (ISA 7.12.2 16-bit 16x32 layout), hoisted per mt
                v16h afr[4];
                const int arow = mt * 16 + mrow;
                #pragma unroll
                for (int kc = 0; kc < 4; ++kc) {
                    v16h f = {};
                    if (arow < 25) {
                        const _Float16* base =
                            &stage[arow / KK][xloc + arow % KK][0];
                        #pragma unroll
                        for (int e = 0; e < 16; ++e) {
                            const int k = kc * 32 + ((e >> 3) << 4) + lhalf * 8
                                        + (((e >> 1) & 3) << 1) + (e & 1);
                            f[e] = base[k];      // contiguous -> ds_load_b128
                        }
                    }
                    afr[kc] = f;
                }
                #pragma unroll
                for (int nt = 0; nt < 2; ++nt) {
                    const int brow = nt * 16 + mrow;   // B[k][n] = A[n][k]
                    v8f acc = {};
                    #pragma unroll
                    for (int kc = 0; kc < 4; ++kc) {
                        v16h bf = {};
                        if (brow < 25) {
                            const _Float16* base =
                                &stage[brow / KK][xloc + brow % KK][0];
                            #pragma unroll
                            for (int e = 0; e < 16; ++e) {
                                const int k = kc * 32 + ((e >> 3) << 4) + lhalf * 8
                                            + (((e >> 1) & 3) << 1) + (e & 1);
                                bf[e] = base[k];
                            }
                        }
                        acc = __builtin_amdgcn_wmma_f32_16x16x32_f16(
                                  false, afr[kc], false, bf,
                                  (short)0, acc, false, false);
                    }
                    // C/D layout: VGPR r -> row r + 8*lhalf, col = lane&15
                    #pragma unroll
                    for (int r = 0; r < 8; ++r)
                        gram[wv][mt * 16 + lhalf * 8 + r][nt * 16 + mrow] = acc[r];
                }
            }
        }
        __syncthreads();

        if (active && lane < 25) {
            const float d = gram[wv][lane][lane];
            nrm[wv][lane] = sqrtf(d > 0.f ? d : 0.f);
        }
        __syncthreads();

        if (active) {
            float sv = -1e30f;
            if (lane < 25) {
                const float np = nrm[wv][lane];
                float a = 0.f;
                #pragma unroll
                for (int q = 0; q < 25; ++q) {
                    float den = np * nrm[wv][q];
                    den = den > 1e-6f ? den : 1e-6f;
                    a += 1.f - gram[wv][lane][q] / den;
                }
                sv = a * (1.f / 25.f);
            }
            srow[wv][lane] = sv;
        }
        __syncthreads();

        if (active && lane == 0) {
            float mm = srow[wv][0];
            #pragma unroll
            for (int q = 1; q < 25; ++q) mm = fmaxf(mm, srow[wv][q]);
            mm = fmaxf(mm, 0.f);     // cosine-distance mean is >= 0
            atomicMax(&bmax[b], __float_as_uint(mm));
        }
        __syncthreads();             // protect gram/nrm/srow WAR for next slot
    }
}

__global__ void init_kernel(unsigned int* bmax)
{
    if (threadIdx.x < BB) bmax[threadIdx.x] = 0u;
}

__global__ void finalize_kernel(const unsigned int* __restrict__ bmax,
                                const int* __restrict__ label,
                                float* __restrict__ out)
{
    if (threadIdx.x == 0) {
        float rs = 0.f, fs = 0.f, rc = 0.f, fc = 0.f;
        for (int i = 0; i < BB; ++i) {
            const float t = __uint_as_float(bmax[i]);
            if (label[i]) { fs += t; fc += 1.f; }
            else          { rs += t; rc += 1.f; }
        }
        const float loss = 1.f - fs / fc + rs / rc;
        out[0] = loss > 0.f ? loss : 0.f;
    }
}

extern "C" void kernel_launch(void* const* d_in, const int* in_sizes, int n_in,
                              void* d_out, int out_size, void* d_ws, size_t ws_size,
                              hipStream_t stream)
{
    const float* feat  = (const float*)d_in[0];
    const int*   label = (const int*)d_in[1];
    unsigned int* bmax = (unsigned int*)d_ws;   // 32 slots

    init_kernel<<<1, 32, 0, stream>>>(bmax);
    gram_cosine_kernel<<<NBLK, WAVES * 32, 0, stream>>>(feat, bmax);
    finalize_kernel<<<1, 32, 0, stream>>>(bmax, label, (float*)d_out);
}
